// PointCloud2Mesh_62388694942113
// MI455X (gfx1250) — matrix operand order repaired and checked by reading, not compiled, not run on hardware
//
#include <hip/hip_runtime.h>
#include <hip/hip_bf16.h>
#include <cstddef>
#include <cstdint>

typedef __attribute__((ext_vector_type(16))) _Float16 v16h;
typedef __attribute__((ext_vector_type(8)))  float    v8f;
typedef __attribute__((ext_vector_type(4)))  unsigned int u32x4;
typedef __attribute__((ext_vector_type(8)))  int      i32x8;
typedef __attribute__((ext_vector_type(4)))  int      i32x4;

#define G 64
#define G3 (G*G*G)            // 262144
#define NB 4
#define NPTS 200000

#if __has_builtin(__builtin_amdgcn_tensor_load_to_lds) && __has_builtin(__builtin_amdgcn_s_wait_tensorcnt)
#define CDNA5_HAS_TDM 1
#else
#define CDNA5_HAS_TDM 0
#endif

// ---------------- workspace layout (bytes) ----------------
#define OFF_VOX   (size_t)0                      //  4 MiB  voxel hist [4,G3] f32
#define OFF_SAMP  ((size_t)4*1024*1024)          //  4 MiB  sampled     [4,G3] f32
#define OFF_OFF   ((size_t)8*1024*1024)          // 12 MiB  offsets [4,3,G3] f32
#define OFF_STATS ((size_t)20*1024*1024)         // 512 B   sum[64],sumsq[64]
#define OFF_SS    (OFF_STATS + 1024)             // 512 B   scale[64],shift[64]
#define OFF_MM    (OFF_STATS + 2048)             // 96 B    per-batch min/max
#define OFF_AF1   (OFF_STATS + 4096)             // 54 KiB  A-fragments conv ow2
#define OFF_AF2   ((size_t)21*1024*1024)         // 54 KiB  A-fragments conv dw2
#define OFF_H     ((size_t)22*1024*1024)         // 128 MiB h_pre [4,64,G3] f16

// low 32 bits of a flat pointer to LDS == byte offset within wave LDS (ISA 10.2)
__device__ __forceinline__ uint32_t lds_byte_offset(const void* p) {
    return (uint32_t)(uintptr_t)p;
}

// ---------------- tiny utility kernels ----------------
__global__ void zero_kernel(float* __restrict__ p, int n) {
    int i = blockIdx.x * 256 + threadIdx.x;
    if (i < n) p[i] = 0.f;
}

// one block per batch: min/max over N points of each of the 3 coords
__global__ __launch_bounds__(256) void minmax_kernel(const float* __restrict__ pts,
                                                     float* __restrict__ mm) {
    __shared__ float smin[3][256];
    __shared__ float smax[3][256];
    int b = blockIdx.x, tid = threadIdx.x;
    float mn[3] = {1e30f, 1e30f, 1e30f}, mx[3] = {-1e30f, -1e30f, -1e30f};
    const float* p = pts + (size_t)b * NPTS * 3;
    for (int i = tid; i < NPTS; i += 256) {
        #pragma unroll
        for (int c = 0; c < 3; ++c) {
            float v = p[i * 3 + c];
            mn[c] = fminf(mn[c], v);
            mx[c] = fmaxf(mx[c], v);
        }
    }
    for (int c = 0; c < 3; ++c) { smin[c][tid] = mn[c]; smax[c][tid] = mx[c]; }
    __syncthreads();
    for (int off = 128; off > 0; off >>= 1) {
        if (tid < off)
            for (int c = 0; c < 3; ++c) {
                smin[c][tid] = fminf(smin[c][tid], smin[c][tid + off]);
                smax[c][tid] = fmaxf(smax[c][tid], smax[c][tid + off]);
            }
        __syncthreads();
    }
    if (tid == 0)
        for (int c = 0; c < 3; ++c) { mm[b * 6 + c] = smin[c][0]; mm[b * 6 + 3 + c] = smax[c][0]; }
}

__global__ void voxelize_kernel(const float* __restrict__ pts,
                                const float* __restrict__ mm,
                                float* __restrict__ hist) {
    int gid = blockIdx.x * 256 + threadIdx.x;
    if (gid >= NB * NPTS) return;
    int b = gid / NPTS;
    const float* p = pts + (size_t)gid * 3;
    int idx[3];
    #pragma unroll
    for (int c = 0; c < 3; ++c) {
        float mn = mm[b * 6 + c], mx = mm[b * 6 + 3 + c];
        float np = (p[c] - mn) / (mx - mn + 1e-6f) * 2.f - 1.f;
        int ii = (int)floorf((np + 1.f) * 0.5f * (float)G);
        idx[c] = min(max(ii, 0), G - 1);
    }
    int lin = (idx[0] * G + idx[1]) * G + idx[2];
    atomicAdd(&hist[(size_t)b * G3 + lin], 1.f);
}

// ---------------- conv 1->64 (k=3^3) via WMMA: M=64(4 waves x16), K=27->32, N=16 voxels ----
// Weights (64x27 f32, contiguous) are staged to LDS by the Tensor Data Mover.
__global__ __launch_bounds__(128) void convA_kernel(const float* __restrict__ vox,
                                                    const float* __restrict__ w,    // [64,27]
                                                    const float* __restrict__ bias, // [64]
                                                    _Float16* __restrict__ out) {   // [4,64,G3] pre-BN
    __shared__ float slab[3 * 3 * 18];
    __shared__ __align__(16) float wsm[64 * 27];
    int tid = threadIdx.x;
    int wave = tid >> 5, lane = tid & 31;
    int wg = blockIdx.x;                       // b*64*64*4 + z*64*4 + y*4 + strip
    int s = wg & 3, y = (wg >> 2) & 63, z = (wg >> 8) & 63, b = wg >> 14;
    int xbase = s * 16;

#if CDNA5_HAS_TDM
    if (tid < 32) {
        // Tensor DMA descriptor (ISA ch.8): 2D tensor 1728x1 f32, tile 1728x1 -> LDS
        unsigned long long ga = (unsigned long long)(uintptr_t)w;
        u32x4 g0;
        g0.x = 1u;                                  // count=1, user descriptor
        g0.y = lds_byte_offset(wsm);                // lds_addr
        g0.z = (unsigned int)ga;                    // global_addr[31:0]
        g0.w = (unsigned int)((ga >> 32) & 0x01FFFFFFull) | 0x80000000u; // addr[56:32], type=2
        i32x8 g1;
        g1[0] = 0x00020000;          // workgroup_mask=0, data_size=2 (4B)
        g1[1] = (int)(1728u << 16);  // tensor_dim0[15:0]=1728 (bits 63:48)
        g1[2] = 0x00010000;          // tensor_dim0 hi=0, tensor_dim1 lo=1
        g1[3] = (int)(1728u << 16);  // tensor_dim1 hi=0, tile_dim0=1728
        g1[4] = 1;                   // tile_dim1=1, tile_dim2=0
        g1[5] = 1728;                // tensor_dim0_stride lo32
        g1[6] = 0;
        g1[7] = 0;
        i32x4 g2 = {0, 0, 0, 0};
        i32x4 g3 = {0, 0, 0, 0};
#if defined(__clang_major__) && (__clang_major__ >= 23)
        i32x8 g4 = {0, 0, 0, 0, 0, 0, 0, 0};
        __builtin_amdgcn_tensor_load_to_lds(g0, g1, g2, g3, g4, 0);
#else
        __builtin_amdgcn_tensor_load_to_lds(g0, g1, g2, g3, 0);
#endif
    }
#endif

    const float* vb = vox + (size_t)b * G3;
    for (int i = tid; i < 162; i += 128) {
        int x = i % 18, dy = (i / 18) % 3, dz = i / 54;
        int gz = z + dz - 1, gy = y + dy - 1, gx = xbase + x - 1;
        float v = 0.f;
        if ((unsigned)gz < (unsigned)G && (unsigned)gy < (unsigned)G && (unsigned)gx < (unsigned)G)
            v = vb[(gz * G + gy) * G + gx];
        slab[i] = v;
    }
#if CDNA5_HAS_TDM
    if (tid < 32) __builtin_amdgcn_s_wait_tensorcnt(0);
#else
    for (int i = tid; i < 64 * 27; i += 128) wsm[i] = w[i];
#endif
    __syncthreads();

    // B fragment: B[k][n], k = tap in 0..26 (pad to 32), n = voxel in strip
    v16h bfrag;
    int n = lane & 15;
    int kh16 = (lane >> 4) << 4;
    #pragma unroll
    for (int i = 0; i < 16; ++i) {
        int k = ((i >> 3) << 4) + ((lane >> 4) << 3) + (i & 7);
        float v = 0.f;
        if (k < 27) {
            int dz = k / 9, rem = k - dz * 9, dy = rem / 3, dx = rem - dy * 3;
            v = slab[(dz * 3 + dy) * 18 + (n + dx)];
        }
        bfrag[i] = (_Float16)v;
    }
    // A fragment per wave: 16 output channels
    v16h afrag;
    int m = wave * 16 + (lane & 15);
    int k8 = (lane >> 4) << 3;
    #pragma unroll
    for (int i = 0; i < 16; ++i) {
        int k = ((i >> 3) << 4) + k8 + (i & 7);
        afrag[i] = (_Float16)((k < 27) ? wsm[m * 27 + k] : 0.f);
    }
    v8f c = {};
    c = __builtin_amdgcn_wmma_f32_16x16x32_f16(false, afrag, false, bfrag, (short)0, c, false, false);

    int rowoff = (lane >> 4) << 3;   // C: M = r + (lane>=16 ? 8 : 0)
    _Float16* ob = out + (size_t)b * 64 * G3 + (size_t)(z * G + y) * G + xbase + n;
    #pragma unroll
    for (int r = 0; r < 8; ++r) {
        int ch = wave * 16 + rowoff + r;
        float v = c[r] + bias[ch];
        ob[(size_t)ch * G3] = (_Float16)v;
    }
}

// ---------------- per-channel BN statistics over [4, G3] ----------------
__global__ __launch_bounds__(256) void bnstats_kernel(const _Float16* __restrict__ h,
                                                      float* __restrict__ stats) {
    int c = blockIdx.x >> 5, blk = blockIdx.x & 31;
    int tid = threadIdx.x;
    float s = 0.f, s2 = 0.f;
    for (int b = 0; b < NB; ++b) {
        const _Float16* p = h + ((size_t)b * 64 + c) * G3 + blk * 8192;
        for (int i = tid; i < 8192; i += 256) {
            float v = (float)p[i];
            s += v; s2 += v * v;
        }
    }
    __shared__ float rs[256], rq[256];
    rs[tid] = s; rq[tid] = s2;
    __syncthreads();
    for (int off = 128; off > 0; off >>= 1) {
        if (tid < off) { rs[tid] += rs[tid + off]; rq[tid] += rq[tid + off]; }
        __syncthreads();
    }
    if (tid == 0) { atomicAdd(&stats[c], rs[0]); atomicAdd(&stats[64 + c], rq[0]); }
}

__global__ void bnfinalize_kernel(const float* __restrict__ stats,
                                  const float* __restrict__ gamma,
                                  const float* __restrict__ beta,
                                  float* __restrict__ ss) {
    int c = threadIdx.x;
    if (c >= 64) return;
    const float Nt = (float)NB * (float)G3;
    float mean = stats[c] / Nt;
    float var  = stats[64 + c] / Nt - mean * mean;
    float sc   = gamma[c] * rsqrtf(var + 1e-5f);
    ss[c] = sc;
    ss[64 + c] = beta[c] - mean * sc;
}

// ------------- pre-swizzle weights [OutC,64,27] into WMMA A-fragment layout -------------
__global__ void afrag_prep(const float* __restrict__ w, _Float16* __restrict__ afrag, int OutC) {
    int idx = blockIdx.x * 256 + threadIdx.x;  // (s2*32 + lane)*16 + i, total 54*32*16
    if (idx >= 54 * 32 * 16) return;
    int i = idx & 15, lane = (idx >> 4) & 31, s2 = idx >> 9;
    int m = lane & 15;
    int k = s2 * 32 + ((i >> 3) << 4) + ((lane >> 4) << 3) + (i & 7);
    float v = 0.f;
    if (m < OutC && k < 1728) {
        int ch = k / 27, tap = k - ch * 27;
        v = w[(m * 64 + ch) * 27 + tap];
    }
    afrag[idx] = (_Float16)v;
}

// ---------------- conv 64->OutC via WMMA: K=1728 in 54 steps --------------------------
// Halo slab is staged raw via async global->LDS copies (ASYNCcnt), then BN+ReLU is
// applied in place in LDS. Rows are 64 elements (128B, 16B aligned); x-borders are
// resolved with a clamped select at fragment-build time.
__global__ __launch_bounds__(128) void convB_kernel(const _Float16* __restrict__ hin, // [4,64,G3]
                                                    const float* __restrict__ ss,     // scale[64],shift[64]
                                                    const _Float16* __restrict__ afrag_g,
                                                    const float* __restrict__ bias,
                                                    float* __restrict__ out,          // [4,OutC,G3]
                                                    int OutC, int act_sigmoid) {
    __shared__ __align__(16) _Float16 slab[32 * 9 * 64];   // 36864 B: 32ch x (3z*3y) x 64x
    int tid = threadIdx.x, lane = tid & 31, wave = tid >> 5;
    int wg = blockIdx.x;                          // b*4096 + z*64 + y
    int y = wg & 63, z = (wg >> 6) & 63, b = wg >> 12;
    const _Float16* hb = hin + (size_t)b * 64 * G3;
    uint32_t slab_base = lds_byte_offset(slab);

    int xbase = wave * 16;
    int n = lane & 15;
    int kh16 = (lane >> 4) << 4;
    v8f c = {};

    for (int g = 0; g < 2; ++g) {
        __syncthreads();
        // async-copy valid halo rows (raw f16): 288 rows x 128B = 2304 x b128 chunks
        for (int t = tid; t < 288 * 8; t += 128) {
            int row = t >> 3, c8 = t & 7;
            int cl = row / 9, rr = row - cl * 9, dz = rr / 3, dy = rr - dz * 3;
            int gz = z + dz - 1, gy = y + dy - 1;
            if ((unsigned)gz < (unsigned)G && (unsigned)gy < (unsigned)G) {
                unsigned long long gaddr =
                    (unsigned long long)(uintptr_t)(hb + (size_t)(g * 32 + cl) * G3 +
                                                    (gz * G + gy) * G) + (unsigned)(c8 * 16);
                uint32_t laddr = slab_base + (uint32_t)(row * 128 + c8 * 16);
                asm volatile("global_load_async_to_lds_b128 %0, %1, off"
                             :: "v"(laddr), "v"(gaddr) : "memory");
            }
        }
        asm volatile("s_wait_asynccnt 0x0" ::: "memory");
        if (g == 0)  // warm L2 for the second channel group (global_prefetch_b8)
            __builtin_prefetch(hb + (size_t)32 * G3 + (size_t)(z * G + y) * G + tid, 0, 1);
        __syncthreads();

        // in-place BN + ReLU; invalid halo rows become exact zeros (SAME padding post-act)
        for (int i = tid; i < 32 * 9 * 64; i += 128) {
            int row = i >> 6;
            int cl = row / 9, rr = row - cl * 9, dz = rr / 3, dy = rr - dz * 3;
            int gz = z + dz - 1, gy = y + dy - 1;
            float v = 0.f;
            if ((unsigned)gz < (unsigned)G && (unsigned)gy < (unsigned)G) {
                int cch = g * 32 + cl;
                v = fmaxf((float)slab[i] * ss[cch] + ss[64 + cch], 0.f);
            }
            slab[i] = (_Float16)v;
        }
        __syncthreads();

        #pragma unroll 3
        for (int s2l = 0; s2l < 27; ++s2l) {
            int s2 = g * 27 + s2l;
            v16h a = *(const v16h*)(afrag_g + (size_t)(s2 * 32 + lane) * 16);
            v16h bm;
            #pragma unroll
            for (int i = 0; i < 16; ++i) {
                int kp = s2l * 32 + ((i >> 3) << 4) + ((lane >> 4) << 3) + (i & 7);          // local k within this 32-channel group
                int cl = kp / 27, tap = kp - cl * 27;
                int dz = tap / 9, rem = tap - dz * 9, dy = rem / 3, dx = rem - dy * 3;
                int gxl = xbase + n + dx - 1;          // -1..64; SAME padding in x
                int idxl = ((cl * 9 + dz * 3 + dy) << 6) + (gxl & 63);
                _Float16 hv = slab[idxl];
                bm[i] = ((unsigned)gxl < (unsigned)G) ? hv : (_Float16)0.f;
            }
            c = __builtin_amdgcn_wmma_f32_16x16x32_f16(false, a, false, bm, (short)0, c, false, false);
        }
    }

    int rowoff = (lane >> 4) << 3;
    size_t outbase = (size_t)b * OutC * G3 + (size_t)(z * G + y) * G + xbase + n;
    #pragma unroll
    for (int r = 0; r < 8; ++r) {
        int m = rowoff + r;
        if (m < OutC) {
            float v = c[r] + bias[m];
            if (act_sigmoid) v = 1.f / (1.f + __expf(-v));
            out[outbase + (size_t)m * G3] = v;
        }
    }
}

// ---------------- grid offset + trilinear border-clamped sampling ----------------
__global__ void gridsample_kernel(const float* __restrict__ vox,
                                  const float* __restrict__ off,
                                  float* __restrict__ out) {
    int gid = blockIdx.x * 256 + threadIdx.x;
    if (gid >= NB * G3) return;
    int b = gid >> 18, v = gid & (G3 - 1);
    int d = v >> 12, h = (v >> 6) & 63, w = v & 63;
    const float step = 2.f / (float)(G - 1);
    float zz = -1.f + d * step, yy = -1.f + h * step, xx = -1.f + w * step;
    const float* ob = off + (size_t)b * 3 * G3;
    // base stacked (z,y,x); grid_sample reads ch0 as x-coordinate (faithful to reference)
    float g0 = fminf(fmaxf(zz + 0.1f * ob[v], -1.f), 1.f);
    float g1 = fminf(fmaxf(yy + 0.1f * ob[G3 + v], -1.f), 1.f);
    float g2 = fminf(fmaxf(xx + 0.1f * ob[2 * G3 + v], -1.f), 1.f);
    float ix = fminf(fmaxf(((g0 + 1.f) * (float)G - 1.f) * 0.5f, 0.f), (float)(G - 1));
    float iy = fminf(fmaxf(((g1 + 1.f) * (float)G - 1.f) * 0.5f, 0.f), (float)(G - 1));
    float iz = fminf(fmaxf(((g2 + 1.f) * (float)G - 1.f) * 0.5f, 0.f), (float)(G - 1));
    float ix0 = floorf(ix), iy0 = floorf(iy), iz0 = floorf(iz);
    float fx = ix - ix0, fy = iy - iy0, fz = iz - iz0;
    const float* vb = vox + (size_t)b * G3;
    float acc = 0.f;
    #pragma unroll
    for (int oz = 0; oz < 2; ++oz)
        #pragma unroll
        for (int oy = 0; oy < 2; ++oy)
            #pragma unroll
            for (int ox = 0; ox < 2; ++ox) {
                int zi = min(max((int)iz0 + oz, 0), G - 1);
                int yi = min(max((int)iy0 + oy, 0), G - 1);
                int xi = min(max((int)ix0 + ox, 0), G - 1);
                float wgt = (oz ? fz : 1.f - fz) * (oy ? fy : 1.f - fy) * (ox ? fx : 1.f - fx);
                acc += vb[(zi * G + yi) * G + xi] * wgt;
            }
    out[gid] = acc;
}

// ---------------- launch ----------------
extern "C" void kernel_launch(void* const* d_in, const int* in_sizes, int n_in,
                              void* d_out, int out_size, void* d_ws, size_t ws_size,
                              hipStream_t stream) {
    const float* points = (const float*)d_in[0];
    const float* ow1    = (const float*)d_in[1];
    const float* ob1    = (const float*)d_in[2];
    const float* ogamma = (const float*)d_in[3];
    const float* obeta  = (const float*)d_in[4];
    const float* ow2    = (const float*)d_in[5];
    const float* ob2    = (const float*)d_in[6];
    const float* dw1    = (const float*)d_in[7];
    const float* db1    = (const float*)d_in[8];
    const float* dgamma = (const float*)d_in[9];
    const float* dbeta  = (const float*)d_in[10];
    const float* dw2    = (const float*)d_in[11];
    const float* db2    = (const float*)d_in[12];

    char* ws = (char*)d_ws;
    float*     vox    = (float*)(ws + OFF_VOX);
    float*     samp   = (float*)(ws + OFF_SAMP);
    float*     offb   = (float*)(ws + OFF_OFF);
    float*     stats  = (float*)(ws + OFF_STATS);
    float*     ss     = (float*)(ws + OFF_SS);
    float*     mm     = (float*)(ws + OFF_MM);
    _Float16*  afragA = (_Float16*)(ws + OFF_AF1);
    _Float16*  afragB = (_Float16*)(ws + OFF_AF2);
    _Float16*  hbuf   = (_Float16*)(ws + OFF_H);

    // voxelize
    zero_kernel<<<(NB * G3 + 255) / 256, 256, 0, stream>>>(vox, NB * G3);
    zero_kernel<<<1, 256, 0, stream>>>(stats, 128);
    minmax_kernel<<<NB, 256, 0, stream>>>(points, mm);
    voxelize_kernel<<<(NB * NPTS + 255) / 256, 256, 0, stream>>>(points, mm, vox);

    // offset branch: conv1 -> BN stats -> conv2(64->3)
    convA_kernel<<<NB * G * G * 4, 128, 0, stream>>>(vox, ow1, ob1, hbuf);
    bnstats_kernel<<<64 * 32, 256, 0, stream>>>(hbuf, stats);
    bnfinalize_kernel<<<1, 64, 0, stream>>>(stats, ogamma, obeta, ss);
    afrag_prep<<<(54 * 32 * 16 + 255) / 256, 256, 0, stream>>>(ow2, afragA, 3);
    convB_kernel<<<NB * G * G, 128, 0, stream>>>(hbuf, ss, afragA, ob2, offb, 3, 0);

    // dynamic grid sampling
    gridsample_kernel<<<(NB * G3 + 255) / 256, 256, 0, stream>>>(vox, offb, samp);

    // decoder: conv1 -> BN stats -> conv2(64->1) + sigmoid
    convA_kernel<<<NB * G * G * 4, 128, 0, stream>>>(samp, dw1, db1, hbuf);
    zero_kernel<<<1, 256, 0, stream>>>(stats, 128);
    bnstats_kernel<<<64 * 32, 256, 0, stream>>>(hbuf, stats);
    bnfinalize_kernel<<<1, 64, 0, stream>>>(stats, dgamma, dbeta, ss);
    afrag_prep<<<(54 * 32 * 16 + 255) / 256, 256, 0, stream>>>(dw2, afragB, 1);
    convB_kernel<<<NB * G * G, 128, 0, stream>>>(hbuf, ss, afragB, db2, (float*)d_out, 1, 1);
}
